// DecoderOnlyBlock_69501160784122
// MI455X (gfx1250) — matrix operand order repaired and checked by reading, not compiled, not run on hardware
//
#include <hip/hip_runtime.h>

#define Bq 4
#define Sq 2048
#define Dq 512
#define Hq 8
#define DKq 64
#define DFFq 2048

typedef __bf16 bf16;
typedef __attribute__((ext_vector_type(16))) __bf16 v16bf;
typedef __attribute__((ext_vector_type(8)))  __bf16 v8bf;
typedef __attribute__((ext_vector_type(8)))  float  v8f;
typedef __attribute__((ext_vector_type(4)))  unsigned int u32x4;
typedef __attribute__((ext_vector_type(8)))  int i32x8;
typedef __attribute__((ext_vector_type(4)))  int i32x4;

static __device__ __forceinline__ v16bf pack16(v8bf lo, v8bf hi) {
  v16bf r;
#pragma unroll
  for (int e = 0; e < 8; ++e) { r[e] = lo[e]; r[e + 8] = hi[e]; }
  return r;
}

// ---- Tensor Data Mover: 2D tile (bf16, 2-byte elements) global -> LDS ----
// Builds a D# per CDNA5 ISA 8.3/8.4 and issues TENSOR_LOAD_TO_LDS.
// group0: [1:0]=count=1, [63:32]=lds byte addr, [120:64]=global addr, [127:126]=type=2
// group1: [17:16]=data_size=1(2B), dims/strides in 2B units
// This toolchain's builtin is the 6-arg form: (g0, g1, g2, g3, g4, cpol).
static __device__ __forceinline__ void tdm_load_2d_bf16(
    unsigned lds_addr, const void* gptr,
    unsigned tile_x, unsigned tile_y,
    unsigned tensor_d0, unsigned tensor_d1,
    unsigned stride_elems)
{
  unsigned long long ga = (unsigned long long)gptr;
  u32x4 g0;
  g0[0] = 1u;                                            // count=1 (valid), user mode
  g0[1] = lds_addr;                                      // LDS byte address
  g0[2] = (unsigned)(ga & 0xffffffffu);                  // global_addr[31:0]
  g0[3] = (unsigned)((ga >> 32) & 0x01ffffffu) | (2u << 30); // global_addr[56:32] | type=2
  i32x8 g1;
  g1[0] = (int)(1u << 16);                               // data_size=1 -> 2 bytes
  g1[1] = (int)((tensor_d0 & 0xffffu) << 16);            // tensor_dim0[15:0]
  g1[2] = (int)((tensor_d0 >> 16) | ((tensor_d1 & 0xffffu) << 16)); // d0 hi | d1 lo
  g1[3] = (int)((tensor_d1 >> 16) | (tile_x << 16));     // d1 hi | tile_dim0
  g1[4] = (int)(tile_y & 0xffffu);                       // tile_dim1, tile_dim2=0
  g1[5] = (int)stride_elems;                             // tensor_dim0_stride[31:0]
  g1[6] = 0;                                             // stride hi | dim1_stride lo
  g1[7] = 0;
  i32x4 z4 = {0, 0, 0, 0};
  i32x8 z8 = {0, 0, 0, 0, 0, 0, 0, 0};
  __builtin_amdgcn_tensor_load_to_lds(g0, g1, z4, z4, z8, 0);
}

static __device__ __forceinline__ unsigned lds_off_of(const void* p) {
  return (unsigned)(unsigned long long)(size_t)p;        // LDS offset = low 32 bits
}

// ---------------- LayerNorm -> bf16 ----------------
__global__ __launch_bounds__(256) void ln_bf16_kernel(
    const float* __restrict__ x, const float* __restrict__ g,
    const float* __restrict__ b, bf16* __restrict__ out)
{
  __shared__ float red[256];
  const int row = blockIdx.x, tid = threadIdx.x;
  const float* xr = x + (size_t)row * Dq;
  float v0 = xr[tid], v1 = xr[tid + 256];
  red[tid] = v0 + v1; __syncthreads();
  for (int st = 128; st > 0; st >>= 1) { if (tid < st) red[tid] += red[tid + st]; __syncthreads(); }
  float mu = red[0] * (1.0f / Dq);
  __syncthreads();
  float d0 = v0 - mu, d1 = v1 - mu;
  red[tid] = d0 * d0 + d1 * d1; __syncthreads();
  for (int st = 128; st > 0; st >>= 1) { if (tid < st) red[tid] += red[tid + st]; __syncthreads(); }
  float rstd = rsqrtf(red[0] * (1.0f / Dq) + 1e-6f);
  bf16* orow = out + (size_t)row * Dq;
  orow[tid]       = (bf16)(d0 * rstd * g[tid] + b[tid]);
  orow[tid + 256] = (bf16)(d1 * rstd * g[tid + 256] + b[tid + 256]);
}

// ---------------- bf16 WMMA GEMM: C = A[MxK](bf16) @ W[KxN](f32->bf16) ----------------
// MODE 0: out bf16 = AB ; MODE 1: out f32 = resid + bias + AB ; MODE 2: out bf16 = relu(AB + bias)
template<int MODE>
__global__ __launch_bounds__(256) void gemm_bf16_kernel(
    const bf16* __restrict__ A, const float* __restrict__ W,
    const float* __restrict__ bias, const float* __restrict__ resid,
    float* __restrict__ outf, bf16* __restrict__ outb,
    int M, int N, int K)
{
  __shared__ bf16 Al[128][32];
  __shared__ bf16 Wt[64][32];   // transposed: Wt[n][k]
  const int tid = threadIdx.x;
  const int lane = tid & 31, wid = tid >> 5;
  const int wm = wid >> 1, wn = wid & 1;
  const int hm = lane >> 4, lx = lane & 15;
  const int m0 = blockIdx.x * 128, n0 = blockIdx.y * 64;
  v8f acc[2][2] = {};
  for (int k0 = 0; k0 < K; k0 += 32) {
    __syncthreads();
    // stage A tile 128x32 via Tensor Data Mover (wave 0 issues one DMA)
    if (wid == 0) {
      tdm_load_2d_bf16(lds_off_of(&Al[0][0]),
                       A + (size_t)m0 * K + k0,
                       /*tile_x=*/32, /*tile_y=*/128,
                       /*tensor_d0=*/32, /*tensor_d1=*/128,
                       /*stride=*/(unsigned)K);
    }
    { // stage W tile 32x64 transposed, f32 -> bf16 (TDM can't convert)
      int n = tid >> 2, kk0 = (tid & 3) * 8;
      const float* wp = W + (size_t)(k0 + kk0) * N + (n0 + n);
#pragma unroll
      for (int kk = 0; kk < 8; ++kk)
        Wt[n][kk0 + kk] = (bf16)wp[(size_t)kk * N];
    }
    if (k0 + 32 < K) {
      int r = tid >> 1, c = (tid & 1) * 16;
      __builtin_prefetch(A + (size_t)(m0 + r) * K + (k0 + 32 + c), 0, 1);
    }
    __builtin_amdgcn_s_wait_tensorcnt(0);   // wave0 drains TDM; no-op elsewhere
    __syncthreads();
    v16bf a[2], bfrag[2];
#pragma unroll
    for (int mi = 0; mi < 2; ++mi) {
      int r = wm * 32 + mi * 16 + lx;
      v8bf lo = *(const v8bf*)&Al[r][hm * 16];
      v8bf hi = *(const v8bf*)&Al[r][hm * 16 + 8];
      a[mi] = pack16(lo, hi);
    }
#pragma unroll
    for (int ni = 0; ni < 2; ++ni) {
      int n = wn * 32 + ni * 16 + lx;
      bfrag[ni] = *(const v16bf*)&Wt[n][hm * 16];
    }
#pragma unroll
    for (int mi = 0; mi < 2; ++mi)
#pragma unroll
      for (int ni = 0; ni < 2; ++ni)
        acc[mi][ni] = __builtin_amdgcn_wmma_f32_16x16x32_bf16(
            false, a[mi], false, bfrag[ni], (short)0, acc[mi][ni], false, false);
  }
#pragma unroll
  for (int mi = 0; mi < 2; ++mi)
#pragma unroll
    for (int ni = 0; ni < 2; ++ni)
#pragma unroll
      for (int r = 0; r < 8; ++r) {
        int row = m0 + wm * 32 + mi * 16 + r + 8 * hm;
        int col = n0 + wn * 32 + ni * 16 + lx;
        float v = acc[mi][ni][r];
        size_t idx = (size_t)row * N + col;
        if (MODE == 0) {
          outb[idx] = (bf16)v;
        } else if (MODE == 1) {
          outf[idx] = resid[idx] + bias[col] + v;
        } else {
          v += bias[col];
          outb[idx] = (bf16)fmaxf(v, 0.0f);
        }
      }
}

// ---------------- Fused causal attention with relative-position skew ----------------
// grid: (S/64, B*H), block 128 (4 waves). Each block: 64 query rows of one (b,h).
__global__ __launch_bounds__(128) void attn_rel_kernel(
    const bf16* __restrict__ qb, const bf16* __restrict__ kb,
    const bf16* __restrict__ vb, const float* __restrict__ relw,
    bf16* __restrict__ outb)
{
  __shared__ bf16 Kl[64][64];        // K tile, row-major (j, d)  -- staged by TDM
  __shared__ bf16 Vt[64][64];        // V tile transposed: Vt[d][jj]
  __shared__ bf16 El[64][64];        // rel-emb tile (r, d)
  __shared__ float QE[2][64][68];    // sliding q·E^T window (two 64-wide tiles)
  __shared__ bf16 Pl[64][64];        // probs (bf16) for P@V re-load as A frags

  const int bh = blockIdx.y, b = bh >> 3, h = bh & 7;
  const int i0 = blockIdx.x * 64;
  const int tid = threadIdx.x, lane = tid & 31, wid = tid >> 5;
  const int hm = lane >> 4, lx = lane & 15;

  const size_t headoff = (size_t)b * Sq * Dq + (size_t)h * DKq;

  // Q fragments (this wave's 16 rows), K-split into two 16x32 frags over d
  v16bf qa[2];
  {
    int s = i0 + wid * 16 + lx;
    const bf16* qr = qb + headoff + (size_t)s * Dq;
#pragma unroll
    for (int kf = 0; kf < 2; ++kf) {
      v8bf lo = *(const v8bf*)(qr + kf * 32 + hm * 16);
      v8bf hi = *(const v8bf*)(qr + kf * 32 + hm * 16 + 8);
      qa[kf] = pack16(lo, hi);
    }
  }

  v8f o[4] = {};
  float mr[8], lr[8];
#pragma unroll
  for (int r = 0; r < 8; ++r) { mr[r] = -1e30f; lr[r] = 0.0f; }

  auto stage_e = [&](int rbase) {
    int rr = tid >> 1, d0 = (tid & 1) * 32;
    int r = rbase + rr;
    if (r < Sq) {
      const float4* er = (const float4*)(relw + ((size_t)h * Sq + r) * DKq + d0);
#pragma unroll
      for (int q4 = 0; q4 < 8; ++q4) {
        float4 f = er[q4];
        El[rr][d0 + q4 * 4 + 0] = (bf16)f.x;
        El[rr][d0 + q4 * 4 + 1] = (bf16)f.y;
        El[rr][d0 + q4 * 4 + 2] = (bf16)f.z;
        El[rr][d0 + q4 * 4 + 3] = (bf16)f.w;
      }
    } else {
#pragma unroll
      for (int d = 0; d < 32; ++d) El[rr][d0 + d] = (bf16)0.0f;
    }
  };

  auto qer_wmma = [&](int buf) {   // QE[buf] = Qrows @ El^T  (64 rows x 64 r)
    v8f qacc[4] = {};
#pragma unroll
    for (int ni = 0; ni < 4; ++ni)
#pragma unroll
      for (int kf = 0; kf < 2; ++kf) {
        v16bf be = *(const v16bf*)&El[ni * 16 + lx][kf * 32 + hm * 16];
        qacc[ni] = __builtin_amdgcn_wmma_f32_16x16x32_bf16(
            false, qa[kf], false, be, (short)0, qacc[ni], false, false);
      }
#pragma unroll
    for (int ni = 0; ni < 4; ++ni)
#pragma unroll
      for (int r = 0; r < 8; ++r)
        QE[buf][wid * 16 + r + 8 * hm][ni * 16 + lx] = qacc[ni][r];
  };

  const int nt = i0 / 64 + 1;   // causal: key tiles j0 = 0..i0
  for (int t = 0; t < nt; ++t) {
    const int j0 = t * 64;
    __syncthreads();
    // stage K tile via TDM (wave 0); V tile transposed by waves 2,3
    if (wid == 0) {
      tdm_load_2d_bf16(lds_off_of(&Kl[0][0]),
                       kb + headoff + (size_t)j0 * Dq,
                       /*tile_x=*/64, /*tile_y=*/64,
                       /*tensor_d0=*/64, /*tensor_d1=*/64,
                       /*stride=*/Dq);
    }
    if (tid >= 64) {
      int jj = tid - 64;
      const bf16* vr = vb + headoff + (size_t)(j0 + jj) * Dq;
#pragma unroll
      for (int d = 0; d < 64; ++d) Vt[d][jj] = vr[d];
    }
    if (t == 0) {               // fill low half of the rel window once
      stage_e(Sq - 64 - i0);
      __syncthreads();
      qer_wmma(0);
      __syncthreads();
    }
    stage_e(Sq - i0 + j0);      // new high tile (guards r >= S with zeros)
    __builtin_amdgcn_s_wait_tensorcnt(0);   // wave0: K tile DMA complete
    __syncthreads();
    qer_wmma((t + 1) & 1);

    // content scores: S = Q @ K^T
    v8f sacc[4] = {};
#pragma unroll
    for (int ni = 0; ni < 4; ++ni)
#pragma unroll
      for (int kf = 0; kf < 2; ++kf) {
        v16bf bk = *(const v16bf*)&Kl[ni * 16 + lx][kf * 32 + hm * 16];
        sacc[ni] = __builtin_amdgcn_wmma_f32_16x16x32_bf16(
            false, qa[kf], false, bk, (short)0, sacc[ni], false, false);
      }
    __syncthreads();            // QE writes visible

    const int blow = t & 1;
    float p[4][8], rmax[8];
#pragma unroll
    for (int r = 0; r < 8; ++r) rmax[r] = -1e30f;
#pragma unroll
    for (int ni = 0; ni < 4; ++ni)
#pragma unroll
      for (int r = 0; r < 8; ++r) {
        int rowl = r + 8 * hm;
        int qrow = wid * 16 + rowl;
        int col  = ni * 16 + lx;
        int idx  = 63 - qrow + col;   // skew gather: r - rwin0, in [0,126]
        float s = sacc[ni][r] * 0.125f
                + QE[(idx >= 64) ? (blow ^ 1) : blow][qrow][idx & 63];
        if (j0 + col > i0 + qrow) s = -1e9f;   // causal mask
        p[ni][r] = s;
        rmax[r] = fmaxf(rmax[r], s);
      }
#pragma unroll
    for (int st = 1; st < 16; st <<= 1)
#pragma unroll
      for (int r = 0; r < 8; ++r)
        rmax[r] = fmaxf(rmax[r], __shfl_xor(rmax[r], st, 32));
    float fs[8], rsum[8];
#pragma unroll
    for (int r = 0; r < 8; ++r) {
      float mnew = fmaxf(mr[r], rmax[r]);
      fs[r] = __expf(mr[r] - mnew);
      mr[r] = mnew;
      float sum = 0.0f;
#pragma unroll
      for (int ni = 0; ni < 4; ++ni) {
        float e = __expf(p[ni][r] - mnew);
        p[ni][r] = e;
        sum += e;
      }
      rsum[r] = sum;
    }
#pragma unroll
    for (int st = 1; st < 16; st <<= 1)
#pragma unroll
      for (int r = 0; r < 8; ++r)
        rsum[r] += __shfl_xor(rsum[r], st, 32);
#pragma unroll
    for (int r = 0; r < 8; ++r) lr[r] = lr[r] * fs[r] + rsum[r];
#pragma unroll
    for (int di = 0; di < 4; ++di)
#pragma unroll
      for (int r = 0; r < 8; ++r) o[di][r] *= fs[r];

    // probs (C layout) -> LDS -> re-load as A fragments (wave-private rows)
#pragma unroll
    for (int ni = 0; ni < 4; ++ni)
#pragma unroll
      for (int r = 0; r < 8; ++r)
        Pl[wid * 16 + r + 8 * hm][ni * 16 + lx] = (bf16)p[ni][r];

    v16bf pa[2];
    {
      const bf16* pr = &Pl[wid * 16 + lx][0];
#pragma unroll
      for (int kf = 0; kf < 2; ++kf) {
        v8bf lo = *(const v8bf*)(pr + kf * 32 + hm * 16);
        v8bf hi = *(const v8bf*)(pr + kf * 32 + hm * 16 + 8);
        pa[kf] = pack16(lo, hi);
      }
    }
    // O += P @ V
#pragma unroll
    for (int di = 0; di < 4; ++di)
#pragma unroll
      for (int kf = 0; kf < 2; ++kf) {
        v16bf bv = *(const v16bf*)&Vt[di * 16 + lx][kf * 32 + hm * 16];
        o[di] = __builtin_amdgcn_wmma_f32_16x16x32_bf16(
            false, pa[kf], false, bv, (short)0, o[di], false, false);
      }
  }

  // normalize and store to [M, D] at column h*DK + d
#pragma unroll
  for (int di = 0; di < 4; ++di)
#pragma unroll
    for (int r = 0; r < 8; ++r) {
      int qrow = wid * 16 + r + 8 * hm;
      size_t row = (size_t)b * Sq + i0 + qrow;
      int col = h * DKq + di * 16 + lx;
      outb[row * Dq + col] = (bf16)(o[di][r] / lr[r]);
    }
}

// ---------------- launch ----------------
extern "C" void kernel_launch(void* const* d_in, const int* in_sizes, int n_in,
                              void* d_out, int out_size, void* d_ws, size_t ws_size,
                              hipStream_t stream)
{
  (void)in_sizes; (void)n_in; (void)out_size; (void)ws_size;
  const float* x    = (const float*)d_in[0];
  // d_in[1] = mask (causal tril) -> implemented analytically in attn kernel
  const float* wq   = (const float*)d_in[2];
  const float* wk   = (const float*)d_in[3];
  const float* wv   = (const float*)d_in[4];
  const float* w0   = (const float*)d_in[5];
  const float* b0   = (const float*)d_in[6];
  const float* relw = (const float*)d_in[7];
  const float* ln1g = (const float*)d_in[8];
  const float* ln1b = (const float*)d_in[9];
  const float* ln2g = (const float*)d_in[10];
  const float* ln2b = (const float*)d_in[11];
  const float* ffw1 = (const float*)d_in[12];
  const float* ffb1 = (const float*)d_in[13];
  const float* ffw2 = (const float*)d_in[14];
  const float* ffb2 = (const float*)d_in[15];

  char* ws = (char*)d_ws;
  const size_t MB = 1ull << 20;
  bf16* hbf   = (bf16*)(ws + 0);         // [8192,512] bf16, 8MB
  bf16* qbuf  = (bf16*)(ws + 8 * MB);
  bf16* kbuf  = (bf16*)(ws + 16 * MB);
  bf16* vbuf  = (bf16*)(ws + 24 * MB);
  bf16* attnb = hbf;                      // alias: hbf dead after QKV GEMMs
  bf16* ffh   = (bf16*)(ws + 0);          // alias [0,32MB): dead by FFN1 time
  float* x1   = (float*)(ws + 32 * MB);   // [8192,512] f32, 16MB
  bf16* h2b   = (bf16*)(ws + 48 * MB);    // 8MB  (total ws use: 56MB)
  float* outp = (float*)d_out;

  const int M = Bq * Sq;
  ln_bf16_kernel<<<dim3(M), dim3(256), 0, stream>>>(x, ln1g, ln1b, hbf);

  dim3 gqkv(M / 128, Dq / 64);
  gemm_bf16_kernel<0><<<gqkv, 256, 0, stream>>>(hbf, wq, nullptr, nullptr, nullptr, qbuf, M, Dq, Dq);
  gemm_bf16_kernel<0><<<gqkv, 256, 0, stream>>>(hbf, wk, nullptr, nullptr, nullptr, kbuf, M, Dq, Dq);
  gemm_bf16_kernel<0><<<gqkv, 256, 0, stream>>>(hbf, wv, nullptr, nullptr, nullptr, vbuf, M, Dq, Dq);

  attn_rel_kernel<<<dim3(Sq / 64, Bq * Hq), dim3(128), 0, stream>>>(qbuf, kbuf, vbuf, relw, attnb);

  gemm_bf16_kernel<1><<<gqkv, 256, 0, stream>>>(attnb, w0, b0, x, x1, nullptr, M, Dq, Dq);
  ln_bf16_kernel<<<dim3(M), dim3(256), 0, stream>>>(x1, ln2g, ln2b, h2b);
  gemm_bf16_kernel<2><<<dim3(M / 128, DFFq / 64), 256, 0, stream>>>(h2b, ffw1, ffb1, nullptr, nullptr, ffh, M, DFFq, Dq);
  gemm_bf16_kernel<1><<<dim3(M / 128, Dq / 64), 256, 0, stream>>>(ffh, ffw2, ffb2, x1, outp, nullptr, M, Dq, DFFq);
}
